// CNF_5557687681565
// MI455X (gfx1250) — matrix operand order repaired and checked by reading, not compiled, not run on hardware
//
#include <hip/hip_runtime.h>

#define NC 30
#define WIDTH 64
#define DD 32
#define BATCH 1048576
#define BLKSZ (WIDTH * DD)

typedef __attribute__((ext_vector_type(16))) _Float16 v16h;
typedef __attribute__((ext_vector_type(2)))  _Float16 h2;
typedef __attribute__((ext_vector_type(8)))  float    v8f;
typedef __attribute__((ext_vector_type(4)))  float    v4f;
typedef __attribute__((ext_vector_type(8)))  unsigned v8u;

// workspace layout (bytes)
#define WS_WTA 0      // 2048 halves: wtA[tile(4)][lane(32)][j(16)]  (A-frag of Wt chunks)
#define WS_UTA 4096   // 2048 halves: utA[mt(2)][q(2)][lane(32)][j(16)] (A-frag, width-permuted (Ut/64)^T)
#define WS_BT  8192   // 64 f32 Bt
#define WS_WU  8448   // 64 f32 wu/64

#define SWZ_SWAPX16 0x401F   // ds_swizzle group-of-32: xor_mask=0x10, and_mask=0x1f

__device__ __forceinline__ float fast_tanh(float x) {
#if __has_builtin(__builtin_amdgcn_tanhf)
  return __builtin_amdgcn_tanhf(x);          // v_tanh_f32 (gfx1250 TRANS op)
#elif __has_builtin(__builtin_amdgcn_tanh_f32)
  return __builtin_amdgcn_tanh_f32(x);
#else
  float e = __expf(2.0f * fabsf(x));         // v_exp_f32, branchless fallback
  float r = 1.0f - 2.0f / (e + 1.0f);
  return copysignf(r, x);
#endif
}

__device__ __forceinline__ float dot30(const float* __restrict__ w, const float* __restrict__ phi) {
  float s = 0.f;
#pragma unroll
  for (int i = 0; i < NC; ++i) s += w[i] * phi[i];
  return s;
}

// ---------------- Phase 1: hypernetwork + fragment packing (1 block) -------
__global__ __launch_bounds__(256) void cnf_hyper(
    const float* __restrict__ t,
    const float* __restrict__ Wc, const float* __restrict__ Wls,
    const float* __restrict__ Wlw, const float* __restrict__ Wlb,
    const float* __restrict__ Uc, const float* __restrict__ Uls,
    const float* __restrict__ Ulw, const float* __restrict__ Ulb,
    const float* __restrict__ Bc, const float* __restrict__ Bls,
    const float* __restrict__ Blw, const float* __restrict__ Blb,
    void* __restrict__ ws) {
  __shared__ float phiW[NC], phiU[NC], phiB[NC];
  __shared__ float Wt[BLKSZ], Ut[BLKSZ], Bts[WIDTH];
  const int tid = threadIdx.x;
  const float tv = t[0];
  if (tid < NC) {
    float d = fabsf(tv - Wc[tid]) / __expf(Wls[tid]);
    phiW[tid] = __expf(-d * d);
  } else if (tid >= 32 && tid < 32 + NC) {
    int i = tid - 32;
    float d = fabsf(tv - Uc[i]) / __expf(Uls[i]);
    phiU[i] = __expf(-d * d);
  } else if (tid >= 64 && tid < 64 + NC) {
    int i = tid - 64;
    float d = fabsf(tv - Bc[i]) / __expf(Bls[i]);
    phiB[i] = __expf(-d * d);
  }
  __syncthreads();
  for (int i = tid; i < BLKSZ; i += 256) {
    Wt[i] = dot30(Wlw + i * NC, phiW) + Wlb[i];
    Ut[i] = dot30(Ulw + i * NC, phiU) + Ulb[i];
  }
  if (tid < WIDTH) Bts[tid] = dot30(Blw + tid * NC, phiB) + Blb[tid];
  __syncthreads();

  float* btw = (float*)((char*)ws + WS_BT);
  float* wuw = (float*)((char*)ws + WS_WU);
  if (tid < WIDTH) {
    float s = 0.f;
#pragma unroll
    for (int d = 0; d < DD; ++d) s += Wt[tid * DD + d] * Ut[tid * DD + d];
    wuw[tid] = s * (1.f / WIDTH);            // fold 1/WIDTH into wu
    btw[tid] = Bts[tid];
  }

  // A-fragment packing here (16xK, contiguous K order, same as zb): lane row m=lane&15; element j -> k = 16*hi+j
  _Float16* wta = (_Float16*)((char*)ws + WS_WTA);
  for (int i = tid; i < 2048; i += 256) {
    int tile = i >> 9, lane = (i >> 4) & 31, j = i & 15;
    int hi = lane >> 4, m = lane & 15;
    int k = 16 * hi + j;
    wta[i] = (_Float16)Wt[(16 * tile + m) * DD + k];                  // A[m][k] = Wt[w][d]
  }
  // GEMM2 A-fragments with the width permutation that makes the GEMM1 D-fragment
  // directly usable as GEMM2 B-fragment (repack (c,r)->(q, 8*(c&1)+r) per lane):
  //   width at contraction index kappa of call q:  w = 32q + kappa
  _Float16* uta = (_Float16*)((char*)ws + WS_UTA);
  for (int i = tid; i < 2048; i += 256) {
    int mt = i >> 10, q = (i >> 9) & 1, lane = (i >> 4) & 31, j = i & 15;
    int hi = lane >> 4, m = lane & 15;
    int kappa = (j < 8) ? (8 * hi + j) : (8 + 8 * hi + j);            // A-layout K map, 0..31
    int w = 32 * q + kappa;
    // fold 1/WIDTH into Ut fragments -> GEMM2 output is final dz
    uta[i] = (_Float16)(Ut[w * DD + 16 * mt + m] * (1.f / WIDTH));
  }
}

// ---------------- Phase 2: batched WMMA flow ------------------------------
#define WPB 8   // waves per 256-thread block
#define TPW 4   // 16-row tiles per wave

__global__ __launch_bounds__(256) void cnf_main(
    const float* __restrict__ z, const void* __restrict__ ws,
    float* __restrict__ out) {
  const int lane = threadIdx.x & 31;
  const int wv   = threadIdx.x >> 5;
  const int hi   = lane >> 4;
  const int n    = lane & 15;          // batch column within tile
  const long gw  = (long)blockIdx.x * WPB + wv;

  const v16h* wtaf = (const v16h*)((const char*)ws + WS_WTA);
  const v16h* utaf = (const v16h*)((const char*)ws + WS_UTA);
  const float* btp = (const float*)((const char*)ws + WS_BT);
  const float* wup = (const float*)((const char*)ws + WS_WU);

  v16h aw[4];
#pragma unroll
  for (int c = 0; c < 4; ++c) aw[c] = wtaf[c * 32 + lane];
  v16h au[2][2];
#pragma unroll
  for (int mt = 0; mt < 2; ++mt)
#pragma unroll
    for (int q = 0; q < 2; ++q) au[mt][q] = utaf[(mt * 2 + q) * 32 + lane];

  // bias as WMMA C-fragment (row w = 16c + 8*hi + r), wu (pre-scaled by 1/WIDTH)
  v8f bfrag[4];
  v8f wuv[4];
#pragma unroll
  for (int c = 0; c < 4; ++c)
#pragma unroll
    for (int r = 0; r < 8; ++r) {
      bfrag[c][r] = btp[16 * c + 8 * hi + r];
      wuv[c][r]   = wup[16 * c + 8 * hi + r];
    }

  float* outlp = out + (long)BATCH * DD;

  for (int tt = 0; tt < TPW; ++tt) {
    const long tile = gw * TPW + tt;
    const long base = tile * 16;

    // z B-fragment: lane col = n, element j -> K(d) = 16*hi + j  (contiguous 64B)
    const v4f* zr = (const v4f*)(z + (base + n) * DD + 16 * hi);
    v4f z0 = __builtin_nontemporal_load(zr + 0);
    v4f z1 = __builtin_nontemporal_load(zr + 1);
    v4f z2 = __builtin_nontemporal_load(zr + 2);
    v4f z3 = __builtin_nontemporal_load(zr + 3);
    v16h zb;
#pragma unroll
    for (int j = 0; j < 4; ++j) {
      zb[j]      = (_Float16)z0[j];
      zb[4 + j]  = (_Float16)z1[j];
      zb[8 + j]  = (_Float16)z2[j];
      zb[12 + j] = (_Float16)z3[j];
    }

    // GEMM1 (bias folded into C): H^T tiles = Wt_chunk(16x32) @ z^T(32x16) + Bt
    v8f acc[4];
#pragma unroll
    for (int c = 0; c < 4; ++c)
      acc[c] = __builtin_amdgcn_wmma_f32_16x16x32_f16(false, aw[c], false, zb,
                                                      (short)0, bfrag[c], false, false);

    // tanh (v_tanh_f32), Jacobian-trace partial, pack h to half2 pairs.
    // Per-lane repack (c,r) -> (q = c>>1, j = 8*(c&1)+r): no cross-lane exchange;
    // the width permutation is absorbed into the phase-1 Ut A-fragments.
    float p = 0.f;
    v8u b0, b1;
#pragma unroll
    for (int c = 0; c < 4; ++c) {
      float hv[8];
#pragma unroll
      for (int r = 0; r < 8; ++r) {
        float x = fast_tanh(acc[c][r]);
        hv[r] = x;
        p += (1.f - x * x) * wuv[c][r];
      }
#pragma unroll
      for (int rr = 0; rr < 4; ++rr) {
        h2 cv;
        cv[0] = (_Float16)hv[2 * rr];
        cv[1] = (_Float16)hv[2 * rr + 1];
        unsigned u = __builtin_bit_cast(unsigned, cv);
        if (c == 0)      b0[rr]     = u;
        else if (c == 1) b0[4 + rr] = u;
        else if (c == 2) b1[rr]     = u;
        else             b1[4 + rr] = u;
      }
    }
    v16h hb0 = __builtin_bit_cast(v16h, b0);
    v16h hb1 = __builtin_bit_cast(v16h, b1);

    // GEMM2: dz^T tiles = perm((Ut/64)^T)_chunk(16x32) @ H^T(32x16), K chained
    const v8f zero = {};
    v8f d0 = __builtin_amdgcn_wmma_f32_16x16x32_f16(false, au[0][0], false, hb0,
                                                    (short)0, zero, false, false);
    d0 = __builtin_amdgcn_wmma_f32_16x16x32_f16(false, au[0][1], false, hb1,
                                                (short)0, d0, false, false);
    v8f d1 = __builtin_amdgcn_wmma_f32_16x16x32_f16(false, au[1][0], false, hb0,
                                                    (short)0, zero, false, false);
    d1 = __builtin_amdgcn_wmma_f32_16x16x32_f16(false, au[1][1], false, hb1,
                                                (short)0, d1, false, false);

    // store dz (already scaled): lane writes d = 16*mt + 8*hi + r for batch row base+n
    float* o = out + (base + n) * DD + 8 * hi;
    v4f s0, s1, s2, s3;
#pragma unroll
    for (int r = 0; r < 4; ++r) {
      s0[r] = d0[r];
      s1[r] = d0[4 + r];
      s2[r] = d1[r];
      s3[r] = d1[4 + r];
    }
    __builtin_nontemporal_store(s0, (v4f*)(o + 0));
    __builtin_nontemporal_store(s1, (v4f*)(o + 4));
    __builtin_nontemporal_store(s2, (v4f*)(o + 16));
    __builtin_nontemporal_store(s3, (v4f*)(o + 20));

    // trace: lane + partner (xor 16) cover all 64 widths for batch row n (wu pre-scaled)
    float tr = p + __uint_as_float(
        (unsigned)__builtin_amdgcn_ds_swizzle((int)__float_as_uint(p), SWZ_SWAPX16));
    if (hi == 0)
      __builtin_nontemporal_store(-tr, outlp + base + n);
  }
}

extern "C" void kernel_launch(void* const* d_in, const int* in_sizes, int n_in,
                              void* d_out, int out_size, void* d_ws, size_t ws_size,
                              hipStream_t stream) {
  const float* t   = (const float*)d_in[0];
  const float* z   = (const float*)d_in[1];
  // d_in[2] logp_z is unused by the reference outputs
  const float* Wc  = (const float*)d_in[3];
  const float* Wls = (const float*)d_in[4];
  const float* Wlw = (const float*)d_in[5];
  const float* Wlb = (const float*)d_in[6];
  const float* Uc  = (const float*)d_in[7];
  const float* Uls = (const float*)d_in[8];
  const float* Ulw = (const float*)d_in[9];
  const float* Ulb = (const float*)d_in[10];
  const float* Bc  = (const float*)d_in[11];
  const float* Bls = (const float*)d_in[12];
  const float* Blw = (const float*)d_in[13];
  const float* Blb = (const float*)d_in[14];

  cnf_hyper<<<1, 256, 0, stream>>>(t, Wc, Wls, Wlw, Wlb,
                                   Uc, Uls, Ulw, Ulb,
                                   Bc, Bls, Blw, Blb, d_ws);

  const int tiles = BATCH / 16;             // 65536
  const int blocks = tiles / (WPB * TPW);   // 2048
  cnf_main<<<blocks, 256, 0, stream>>>(z, d_ws, (float*)d_out);
}
